// sv2el_317827580645
// MI455X (gfx1250) — matrix operand order repaired and checked by reading, not compiled, not run on hardware
//
#include <hip/hip_runtime.h>
#include <hip/hip_bf16.h>
#include <cstdint>

// B=4, N=512, M=32, D=64, H=64
#define NB 4
#define NN 512
#define NM 32
#define ND 64
#define NH 64
#define NT 32            // N / 16 tiles
#define TILE_ELEMS 32768 // 16 i * 64 h * 32 m  (one (b,it) V-tile, f16 elements)
#define QELEMS 8192      // one h-quarter slab: 16 h * 16 i * 32 m f16 elements (16 KB)

typedef __attribute__((ext_vector_type(16))) _Float16 v16h;
typedef __attribute__((ext_vector_type(8)))  float    v8f;
typedef __attribute__((ext_vector_type(4)))  unsigned int u32x4;
typedef __attribute__((ext_vector_type(8)))  int      i32x8;
typedef __attribute__((ext_vector_type(4)))  int      i32x4;

__device__ __forceinline__ v8f wmma16(v16h a, v16h b, v8f c) {
  // emits v_wmma_f32_16x16x32_f16
  return __builtin_amdgcn_wmma_f32_16x16x32_f16(false, a, false, b, (short)0, c, false, false);
}

// A-fragment K index for 16-bit 16x32 A layout:
// VGPR r: lanes 0-15 K = {0..7,16..23}, lanes 16-31 K = +8
__device__ __forceinline__ int a_kidx(int r, int g) {
  return ((r < 4) ? 2 * r : 16 + 2 * (r - 4)) + 8 * g;
}

// LDS byte offset of a shared-memory pointer (flat aperture: low 32 bits = LDS offset)
__device__ __forceinline__ unsigned int lds_off(const void* p) {
  return (unsigned int)(uintptr_t)p;
}

// TDM: 1-D copy of n f16 elements from global to LDS (D# per ISA 8.3/8.4).
// 6-arg builtin form (clang-23 / therock-10.0 headers).
__device__ __forceinline__ void tdm_load_1d(unsigned int lds_addr, const void* gptr,
                                            unsigned int n) {
  const unsigned long long ga = (unsigned long long)(uintptr_t)gptr;
  u32x4 g0;
  g0[0] = 1u;                                   // count=1 (valid), user mode, no gather
  g0[1] = lds_addr;                             // lds_addr (bytes)
  g0[2] = (unsigned int)ga;                     // global_addr[31:0]  -> bits 95:64
  g0[3] = (unsigned int)(ga >> 32) | (2u << 30);// global_addr[56:32] + type=2
  i32x8 g1;
  g1[0] = (int)(1u << 16);                      // data_size = 1 (2 bytes); wg_mask=0
  g1[1] = (int)((n & 0xFFFFu) << 16);           // tensor_dim0[15:0] -> bits 63:48
  g1[2] = (int)((n >> 16) | (1u << 16));        // tensor_dim0[31:16], tensor_dim1 = 1
  g1[3] = (int)((n & 0xFFFFu) << 16);           // tile_dim0 -> bits 127:112
  g1[4] = 0;                                    // tile_dim1 = 0 (unused), tile_dim2 = 0
  g1[5] = (int)n;                               // tensor_dim0_stride[31:0]
  g1[6] = 0;
  g1[7] = 0;
  i32x4 z4 = {0, 0, 0, 0};
  i32x8 z8 = {0, 0, 0, 0, 0, 0, 0, 0};
  __builtin_amdgcn_tensor_load_to_lds(g0, g1, z4, z4, z8, 0);
}

// -------------------------------------------------------------------------
// Kernel 1: V projections with WMMA.
// V1t/V2t layout: [b*32+it][h][i][m] f16, tile = 32768 elems contiguous.
// grid = 256 blocks (128 for V1, 128 for V2), 256 threads.
// -------------------------------------------------------------------------
__global__ __launch_bounds__(256) void proj_v_kernel(
    const float* __restrict__ v1, const float* __restrict__ v0,
    const float* __restrict__ Wv1, const float* __restrict__ Wv2,
    _Float16* __restrict__ V1t, _Float16* __restrict__ V2t) {
  const int bid   = blockIdx.x;
  const int which = bid >> 7;          // 0 -> V1 (from v1), 1 -> V2 (from v0)
  const int tb    = bid & 127;         // b*32 + it
  const float* __restrict__ src = which ? v0 : v1;
  const float* __restrict__ W   = which ? Wv2 : Wv1;
  _Float16* __restrict__ dst    = (which ? V2t : V1t) + (size_t)tb * TILE_ELEMS;

  const int lane = threadIdx.x & 31;
  const int wave = threadIdx.x >> 5;
  const int l15  = lane & 15;
  const int g    = lane >> 4;

  // Stage W transposed into LDS: Wt[h][d] so B-fragment K-pairs are contiguous.
  __shared__ __align__(16) _Float16 Wt[64 * 64];            // 8 KB
  __shared__ __align__(16) _Float16 stage[TILE_ELEMS];      // 64 KB output staging
  for (int e = threadIdx.x; e < 4096; e += 256) {
    int h = e >> 6, d = e & 63;
    Wt[e] = (_Float16)W[d * 64 + h];
  }
  __syncthreads();

  // Precompute all B fragments (4 col tiles x 2 K-steps) -- shared by all row tiles.
  v16h bf[4][2];
#pragma unroll
  for (int ct = 0; ct < 4; ++ct) {
    const _Float16* pb = &Wt[(ct * 16 + l15) * 64];
#pragma unroll
    for (int r = 0; r < 8; ++r) {
      int k = a_kidx(r, g);
      bf[ct][0][2 * r] = pb[k];       bf[ct][0][2 * r + 1] = pb[k + 1];
      bf[ct][1][2 * r] = pb[32 + k];  bf[ct][1][2 * r + 1] = pb[33 + k];
    }
  }

  // Rows of this tile: local row = i*32 + m, 512 rows total, 32 row tiles.
  const size_t rowbase = ((size_t)tb * 16) * 32 * 64; // elem offset of (b, i0, m=0, d=0)
#pragma unroll
  for (int rti = 0; rti < 4; ++rti) {
    const int rt = wave * 4 + rti;
    const float* pa = src + rowbase + (size_t)(rt * 16 + l15) * 64;
    v16h a0, a1;
#pragma unroll
    for (int r = 0; r < 8; ++r) {
      int d = a_kidx(r, g);
      a0[2 * r] = (_Float16)pa[d];      a0[2 * r + 1] = (_Float16)pa[d + 1];
      a1[2 * r] = (_Float16)pa[32 + d]; a1[2 * r + 1] = (_Float16)pa[33 + d];
    }
#pragma unroll
    for (int ct = 0; ct < 4; ++ct) {
      v8f acc = {0.f, 0.f, 0.f, 0.f, 0.f, 0.f, 0.f, 0.f};
      acc = wmma16(a0, bf[ct][0], acc);
      acc = wmma16(a1, bf[ct][1], acc);
#pragma unroll
      for (int v = 0; v < 8; ++v) {
        int row = rt * 16 + v + 8 * g;   // D layout: M = v + 8*(lane>=16)
        int i = row >> 5, m = row & 31;
        int h = ct * 16 + l15;           // N = lane & 15
        stage[(h * 16 + i) * 32 + m] = (_Float16)acc[v];
      }
    }
  }
  __syncthreads();
  // Coalesced b128 writeout of the whole 64 KB tile.
  const uint4* s4 = (const uint4*)stage;
  uint4* d4 = (uint4*)dst;
#pragma unroll
  for (int q = 0; q < 16; ++q) d4[threadIdx.x + q * 256] = s4[threadIdx.x + q * 256];
}

// -------------------------------------------------------------------------
// Kernel 2: s1/s2 projections (small, fp32 VALU) + Wm1/Wm2 transposed f16.
// grid = 1024 blocks x 256 threads.
// -------------------------------------------------------------------------
__global__ __launch_bounds__(256) void proj_s_kernel(
    const float* __restrict__ s,
    const float* __restrict__ Ws1, const float* __restrict__ bs1,
    const float* __restrict__ Ws2, const float* __restrict__ bs2,
    const float* __restrict__ Wm1, const float* __restrict__ Wm2,
    float* __restrict__ s1w, float* __restrict__ s2w,
    _Float16* __restrict__ w1tG, _Float16* __restrict__ w2tG) {
  const int tid = blockIdx.x * 256 + threadIdx.x;   // 0 .. 262143
  const int which = tid >> 17;                      // 0 -> s1, 1 -> s2
  const int ee = tid & 131071;
  const int n = ee >> 6, h = ee & 63;
  const float* W = which ? Ws2 : Ws1;
  const float* bias = which ? bs2 : bs1;
  float* out = which ? s2w : s1w;
  float acc = bias[h];
  const float* sr = s + (size_t)n * 64;
#pragma unroll
  for (int d = 0; d < 64; ++d) acc += sr[d] * W[d * 64 + h];
  out[ee] = acc;

  if (blockIdx.x < 32) {
    int idx = blockIdx.x * 256 + threadIdx.x;       // 0 .. 8191
    int wsel = idx >> 12;
    int r = idx & 4095;
    int nn = r >> 6, k = r & 63;                    // wXt[n][k] = Wm[k][n]
    const float* Wm = wsel ? Wm2 : Wm1;
    _Float16* dstt = wsel ? w2tG : w1tG;
    dstt[r] = (_Float16)Wm[k * 64 + nn];
  }
}

// -------------------------------------------------------------------------
// Kernel 3: fused pair + scale + MLP per 16x16 (i,j) tile.
// grid = B*NT*NT = 4096 blocks, 256 threads (8 waves).
// V tiles stream in via TDM (tensor_load_to_lds) double-buffered by h-quarter.
// -------------------------------------------------------------------------
__global__ __launch_bounds__(256) void fused_kernel(
    const _Float16* __restrict__ V1t, const _Float16* __restrict__ V2t,
    const float* __restrict__ s1w, const float* __restrict__ s2w,
    const _Float16* __restrict__ w1tG, const _Float16* __restrict__ w2tG,
    const float* __restrict__ bm1, const float* __restrict__ bm2,
    float* __restrict__ out) {
  const int bid = blockIdx.x;
  const int b  = bid >> 10;
  const int rr = bid & 1023;
  const int it = rr >> 5;
  const int jt = rr & 31;

  // double-buffered V quarter slabs: [buf][V1/V2][16h][16i][32m]  (64 KB)
  __shared__ __align__(16) _Float16 sVbuf[2][2][QELEMS];
  __shared__ __align__(16) _Float16 xls[256 * 64];  // x rows (i*16+j) x h (32 KB)
  __shared__ __align__(16) _Float16 w1t[4096];      // Wm1^T [n][k] (8 KB)
  __shared__ __align__(16) _Float16 w2t[4096];      // Wm2^T (8 KB)
  __shared__ __align__(16) float ss1[16 * 64];      // s1 tile (4 KB)
  __shared__ __align__(16) float ss2[16 * 64];      // s2 tile (4 KB)
  _Float16* hid = &sVbuf[0][0][0];                  // reuse buf0 (32 KB) for MLP hidden

  const int lane = threadIdx.x & 31;
  const int wave = threadIdx.x >> 5;
  const int l15  = lane & 15;
  const int g    = lane >> 4;

  // ---- one-time staging: weights + s tiles (1 uint4 per thread each) ----
  ((uint4*)w1t)[threadIdx.x] = ((const uint4*)w1tG)[threadIdx.x];
  ((uint4*)w2t)[threadIdx.x] = ((const uint4*)w2tG)[threadIdx.x];
  ((uint4*)ss1)[threadIdx.x] = ((const uint4*)(s1w + ((size_t)b * NN + it * 16) * 64))[threadIdx.x];
  ((uint4*)ss2)[threadIdx.x] = ((const uint4*)(s2w + ((size_t)b * NN + jt * 16) * 64))[threadIdx.x];

  const _Float16* tA = V1t + (size_t)(b * 32 + it) * TILE_ELEMS;
  const _Float16* tB = V2t + (size_t)(b * 32 + jt) * TILE_ELEMS;

  // ---- prologue: wave 0 kicks off quarter 0 DMA ----
  if (threadIdx.x < 32) {
    tdm_load_1d(lds_off(&sVbuf[0][0][0]), tA, QELEMS);
    tdm_load_1d(lds_off(&sVbuf[0][1][0]), tB, QELEMS);
  }

  // ---- pair WMMA over four h-quarters, scale, write x to LDS ----
  for (int q = 0; q < 4; ++q) {
    if (threadIdx.x < 32) {
      if (q < 3) { // issue next quarter into the other buffer, then wait current
        tdm_load_1d(lds_off(&sVbuf[(q + 1) & 1][0][0]), tA + (q + 1) * QELEMS, QELEMS);
        tdm_load_1d(lds_off(&sVbuf[(q + 1) & 1][1][0]), tB + (q + 1) * QELEMS, QELEMS);
        __builtin_amdgcn_s_wait_tensorcnt(2);   // current quarter's 2 ops complete
      } else {
        __builtin_amdgcn_s_wait_tensorcnt(0);
      }
    }
    __syncthreads();

    const _Float16* sVa = &sVbuf[q & 1][0][0];
    const _Float16* sVb = &sVbuf[q & 1][1][0];
#pragma unroll
    for (int t = 0; t < 2; ++t) {
      const int hh = wave * 2 + t;               // local h within quarter (16 per q)
      const _Float16* ra = &sVa[(hh * 16 + l15) * 32];  // A row i = l15
      const _Float16* rb = &sVb[(hh * 16 + l15) * 32];  // B col j = l15
      v16h a, bfr;
#pragma unroll
      for (int r = 0; r < 8; ++r) {
        int ma = a_kidx(r, g);                   // A: K over m
        a[2 * r] = ra[ma];  a[2 * r + 1] = ra[ma + 1];
        int mb = a_kidx(r, g);                   // B: K over m
        bfr[2 * r] = rb[mb]; bfr[2 * r + 1] = rb[mb + 1];
      }
      v8f c = {0.f, 0.f, 0.f, 0.f, 0.f, 0.f, 0.f, 0.f};
      c = wmma16(a, bfr, c);                     // pair[i,j] for this h
      const int hg = q * 16 + hh;
      const float s2v = ss2[l15 * 64 + hg];      // j = l15
#pragma unroll
      for (int v = 0; v < 8; ++v) {
        int i = v + 8 * g;
        float x = c[v] * ss1[i * 64 + hg] * s2v;
        xls[(i * 16 + l15) * 64 + hg] = (_Float16)x;
      }
    }
    __syncthreads();
  }

  // ---- MLP layer 1 (X[256x64] @ Wm1 + b, ReLU) ----
#pragma unroll
  for (int rti = 0; rti < 2; ++rti) {
    const int rt = wave * 2 + rti;
#pragma unroll
    for (int ct = 0; ct < 4; ++ct) {
      v8f acc = {0.f, 0.f, 0.f, 0.f, 0.f, 0.f, 0.f, 0.f};
#pragma unroll
      for (int ks = 0; ks < 2; ++ks) {
        const int k0 = ks * 32;
        const _Float16* pa = &xls[(rt * 16 + l15) * 64 + k0];
        const _Float16* pw = &w1t[(ct * 16 + l15) * 64 + k0];
        v16h a, bfr;
#pragma unroll
        for (int r = 0; r < 8; ++r) {
          int ka = a_kidx(r, g);
          a[2 * r] = pa[ka];  a[2 * r + 1] = pa[ka + 1];
          int kb = a_kidx(r, g);
          bfr[2 * r] = pw[kb]; bfr[2 * r + 1] = pw[kb + 1];
        }
        acc = wmma16(a, bfr, acc);
      }
      const float bias = bm1[ct * 16 + l15];
#pragma unroll
      for (int v = 0; v < 8; ++v) {
        float hv = acc[v] + bias;
        hv = hv > 0.f ? hv : 0.f;                // ReLU
        hid[(rt * 16 + v + 8 * g) * 64 + ct * 16 + l15] = (_Float16)hv;
      }
    }
  }
  __syncthreads();

  // ---- MLP layer 2, write output ----
#pragma unroll
  for (int rti = 0; rti < 2; ++rti) {
    const int rt = wave * 2 + rti;
#pragma unroll
    for (int ct = 0; ct < 4; ++ct) {
      v8f acc = {0.f, 0.f, 0.f, 0.f, 0.f, 0.f, 0.f, 0.f};
#pragma unroll
      for (int ks = 0; ks < 2; ++ks) {
        const int k0 = ks * 32;
        const _Float16* pa = &hid[(rt * 16 + l15) * 64 + k0];
        const _Float16* pw = &w2t[(ct * 16 + l15) * 64 + k0];
        v16h a, bfr;
#pragma unroll
        for (int r = 0; r < 8; ++r) {
          int ka = a_kidx(r, g);
          a[2 * r] = pa[ka];  a[2 * r + 1] = pa[ka + 1];
          int kb = a_kidx(r, g);
          bfr[2 * r] = pw[kb]; bfr[2 * r + 1] = pw[kb + 1];
        }
        acc = wmma16(a, bfr, acc);
      }
      const int col = ct * 16 + l15;
      const float bias = bm2[col];
#pragma unroll
      for (int v = 0; v < 8; ++v) {
        int row = rt * 16 + v + 8 * g;           // row = i*16 + j
        int i = row >> 4, j = row & 15;
        size_t gidx = (((size_t)b * NN + it * 16 + i) * NN + (jt * 16 + j)) * (size_t)NH + col;
        out[gidx] = acc[v] + bias;
      }
    }
  }
}

// -------------------------------------------------------------------------
extern "C" void kernel_launch(void* const* d_in, const int* in_sizes, int n_in,
                              void* d_out, int out_size, void* d_ws, size_t ws_size,
                              hipStream_t stream) {
  const float* s   = (const float*)d_in[0];
  const float* v1  = (const float*)d_in[1];
  const float* v0  = (const float*)d_in[2];
  const float* Ws1 = (const float*)d_in[3];
  const float* bs1 = (const float*)d_in[4];
  const float* Ws2 = (const float*)d_in[5];
  const float* bs2 = (const float*)d_in[6];
  const float* Wv1 = (const float*)d_in[7];
  const float* Wv2 = (const float*)d_in[8];
  const float* Wm1 = (const float*)d_in[9];
  const float* bm1 = (const float*)d_in[10];
  const float* Wm2 = (const float*)d_in[11];
  const float* bm2 = (const float*)d_in[12];
  float* out = (float*)d_out;

  char* ws = (char*)d_ws;
  _Float16* V1t  = (_Float16*)(ws);                    // 8 MB
  _Float16* V2t  = (_Float16*)(ws + (8u << 20));       // 8 MB
  float*    s1w  = (float*)   (ws + (16u << 20));      // 512 KB
  float*    s2w  = (float*)   (ws + (16u << 20) + (512u << 10));
  _Float16* w1tG = (_Float16*)(ws + (17u << 20));      // 8 KB
  _Float16* w2tG = (_Float16*)(ws + (17u << 20) + 8192);

  proj_v_kernel<<<256, 256, 0, stream>>>(v1, v0, Wv1, Wv2, V1t, V2t);
  proj_s_kernel<<<1024, 256, 0, stream>>>(s, Ws1, bs1, Ws2, bs2, Wm1, Wm2,
                                          s1w, s2w, w1tG, w2tG);
  fused_kernel<<<NB * NT * NT, 256, 0, stream>>>(V1t, V2t, s1w, s2w,
                                                 w1tG, w2tG, bm1, bm2, out);
}
